// EvalEig_73839077753019
// MI455X (gfx1250) — compile-verified, matrix operand order checked by reading, not run on hardware
//
#include <hip/hip_runtime.h>
#include <hip/hip_bf16.h>

// ---------------- problem constants (from reference) ----------------
#define NE     4096
#define LDIM   3
#define NC     (NE * LDIM)        // 12288 independent (energy, l) columns
#define RNPTS  1000
#define MATCH  100                // int(RN * MATCH_FRAC)
#define NINF   (RNPTS - MATCH)    // 900
#define RMF    100.0f
#define RDF    0.1f               // RM / RN
#define PARA1F 1.0f

typedef __attribute__((ext_vector_type(16))) _Float16 v16h;
typedef __attribute__((ext_vector_type(8)))  float    v8f;
typedef __attribute__((ext_vector_type(2)))  float    v2f;

// Probe: native f32 16x16x4 WMMA (CDNA5 VOP3P op 93). Fall back to the
// codegen-confirmed f16 path if the builtin is not declared.
#if defined(__has_builtin)
#  if __has_builtin(__builtin_amdgcn_wmma_f32_16x16x4_f32)
#    define HAVE_WMMA_F32X4 1
#  endif
#endif

// factor[j] = l(l+1)/r^2 + PARA1*(-e - 1/r),  r = RD*(j+1)
__device__ __forceinline__ float factor_at(int j, float e, float lf) {
    float r    = RDF * (float)(j + 1);
    float invr = 1.0f / r;
    return lf * invr * invr + PARA1F * (-e - invr);
}

__device__ __forceinline__ float powi(float x, int n) {
    float r = 1.0f;
    for (int i = 0; i < n; ++i) r *= x;
    return r;
}

// weighted integrand term of the reference's _integrate(u^2):
// u^4 gets Boole weights (7/14/32), u^2 gets the (faithful) 12*g2 term.
__device__ __forceinline__ float integ_term(float u, int j, int last) {
    if (j > last) return 0.0f;
    float u2 = u * u;
    int   r  = j & 3;
    float w4 = 0.0f, w2 = 0.0f;
    if (r == 0)      w4 = (j == 0 || j == last) ? 7.0f : 14.0f;
    else if (r == 2) w2 = 12.0f;
    else             w4 = 32.0f;
    return (2.0f * RDF / 45.0f) * (w4 * u2 * u2 + w2 * u2);
}

// ---------------------------------------------------------------------------
// K1: one thread per (energy, l) column. Serial Numerov recurrence in
// registers; per-step stores are lane-contiguous (column index == lane) so
// every step emits one fully coalesced global_store_b32 wavefront.
// ws layout (floats): [0]=lfunc_in  [1]=deriv_out_raw (pre energy-flip)
//                     [2]=u_zero[-1] [3]=u_infty[0]  [4]=integ_in [5]=integ_out
// ---------------------------------------------------------------------------
__global__ void __launch_bounds__(128)
solver_kernel(const float* __restrict__ init_energy,
              float* __restrict__ out, float* __restrict__ ws) {
    int c = blockIdx.x * blockDim.x + threadIdx.x;
    if (c >= NC) return;
    int ie = c / LDIM;
    int l  = c - ie * LDIM;
    float e  = init_energy[ie];
    float lf = (float)(l * (l + 1));

    float* u_zero = out + NC;              // [MATCH][NC]
    float* u_inf  = out + NC + MATCH * NC; // [NINF ][NC]

    const float c3  = (3.0f / 40.0f) * RDF * RDF;
    const float c13 = (13.0f / 15.0f) * RDF * RDF;
    const float c7  = (7.0f / 60.0f) * RDF * RDF;

    // ---------------- outward solution u_zero ----------------
    float w0, w1, w2, w3, w4;
    {
        float ww[5];
        for (int i = 0; i < 5; ++i) {
            float r0  = RDF * (float)(i + 1);
            float rl1 = powi(r0, l + 1);
            ww[i] = rl1 - rl1 * r0 / (2.0f * (float)(l + 1)) * PARA1F;
            u_zero[i * NC + c] = ww[i];
        }
        u_zero[5 * NC + c] = ww[4];        // duplicated match sample (uz[:,4][0])
        w0 = ww[0]; w1 = ww[1]; w2 = ww[2]; w3 = ww[3]; w4 = ww[4];
    }
    for (int s = 1; s <= MATCH - 6; ++s) { // 94 steps
        float f0 = factor_at(s,     e, lf);
        float f1 = factor_at(s + 1, e, lf);
        float f2 = factor_at(s + 2, e, lf);
        float f3 = factor_at(s + 3, e, lf);
        float f4 = factor_at(s + 4, e, lf);
        float inv = 1.0f / (1.0f - c3 * f4);
        float a  = -1.0f + c3  * f0 * inv;
        float b  =  2.0f + c13 * f1 * inv;
        float cc = -2.0f + c7  * f2 * inv;
        float d  =  2.0f + c13 * f3 * inv;
        float nw = a * w1 + b * w2 + cc * w3 + d * w4;
        w0 = w1; w1 = w2; w2 = w3; w3 = w4; w4 = nw;
        u_zero[(s + 5) * NC + c] = nw;     // rows 6..99
    }
    // window is now x94..x98; 5-point one-sided derivative at the match point
    float dz = (25.0f * w4 - 48.0f * w3 + 36.0f * w2 - 16.0f * w1 + 3.0f * w0)
               / (12.0f * RDF);
    ws[0 * NC + c] = dz / w4;   // lfunc_in
    ws[2 * NC + c] = w4;        // u_zero[-1]

    // ---------------- inward solution u_infty ----------------
    {
        float sq   = sqrtf(fabsf(e));
        float fact = (l == 0) ? 1.0f : (l == 1 ? 3.0f : 15.0f);
        float ww[5];
        for (int i = 0; i < 5; ++i) {
            float ri   = RMF - (float)(4 - i) * RDF;
            float x    = ri * sq;
            float base = powi(x, l) / fact;
            float x2h  = x * x * 0.5f;
            float t1   = x2h / (float)(2 * l + 3);
            float t2   = x2h * x2h / (2.0f * (float)((2 * l + 3) * (2 * l + 5)));
            ww[i] = ri * base * (1.0f + t1 + t2);
        }
        w0 = ww[0]; w1 = ww[1]; w2 = ww[2]; w3 = ww[3]; w4 = ww[4];
    }
    u_inf[899 * NC + c] = w4;              // y_4 -> row 899 (reversed tail)
    for (int s = 1; s <= NINF - 6; ++s) {  // 894 steps on reversed factor
        float f0 = factor_at(999 - s,       e, lf);
        float f1 = factor_at(999 - (s + 1), e, lf);
        float f2 = factor_at(999 - (s + 2), e, lf);
        float f3 = factor_at(999 - (s + 3), e, lf);
        float f4 = factor_at(999 - (s + 4), e, lf);
        float inv = 1.0f / (1.0f - c3 * f4);
        float a  = -1.0f + c3  * f0 * inv;
        float b  =  2.0f + c13 * f1 * inv;
        float cc = -2.0f + c7  * f2 * inv;
        float d  =  2.0f + c13 * f3 * inv;
        float nw = a * w1 + b * w2 + cc * w3 + d * w4;
        w0 = w1; w1 = w2; w2 = w3; w3 = w4; w4 = nw;
        int k = 4 + s;                     // y_k
        u_inf[(899 - s) * NC + c] = nw;    // reversed segment rows 898..5
        if (k >= 894) u_inf[(k - 894) * NC + c] = nw;  // head rows 0..4
    }
    // window is y894..y898; derivative of reversed u_infty (pre energy-axis flip)
    float dout = (25.0f * w0 - 48.0f * w1 + 36.0f * w2 - 16.0f * w3 + 3.0f * w4)
                 / (12.0f * RDF);
    ws[1 * NC + c] = dout;
    ws[3 * NC + c] = w0;        // u_infty[0]
}

// ---------------------------------------------------------------------------
// K2: WMMA-based batched integration. One wave32 per 16-row tile (768 tiles).
// Preferred path: V_WMMA_F32_16X16X4_F32 — full f32, K=100/900 are exact
// multiples of 4 so no padding and no scaling. B = all-ones turns the matmul
// into an exact per-row reduction (every D column carries the row sum).
// A layout (32-bit 16x4): lanes 0-15 hold K={0,1}, lanes 16-31 hold K={2,3}.
// Data is L2-resident (49 MB << 192 MB L2), so this pass costs no HBM reads.
// ---------------------------------------------------------------------------
__global__ void __launch_bounds__(256)
integrate_kernel(const float* __restrict__ out, float* __restrict__ ws) {
    int wave    = (int)(blockIdx.x * blockDim.x + threadIdx.x) >> 5;
    int lane    = threadIdx.x & 31;
    int rowbase = wave * 16;
    int myrow   = rowbase + (lane & 15);

    const float* u_zero = out + NC;
    const float* u_inf  = out + NC + MATCH * NC;

#if HAVE_WMMA_F32X4
    int kbase = (lane >> 4) * 2;        // lanes 0-15: K 0,1 ; lanes 16-31: K 2,3
    v2f ones2; ones2[0] = 1.0f; ones2[1] = 1.0f;

    // ----- integ_in : K = 100 = 25 chunks of 4 (exact) -----
    v8f acc = {};
    for (int ch = 0; ch < 25; ++ch) {
        int j0 = ch * 4 + kbase;
        v2f a;
        a[0] = integ_term(u_zero[(j0    ) * NC + myrow], j0,     96);
        a[1] = integ_term(u_zero[(j0 + 1) * NC + myrow], j0 + 1, 96);
        acc = __builtin_amdgcn_wmma_f32_16x16x4_f32(
                  false, a, false, ones2, (short)0, acc, false, false);
    }
    if (lane == 0) {
        #pragma unroll
        for (int p = 0; p < 8; ++p) ws[4 * NC + rowbase + p] = acc[p];
    } else if (lane == 16) {
        #pragma unroll
        for (int p = 0; p < 8; ++p) ws[4 * NC + rowbase + 8 + p] = acc[p];
    }

    // ----- integ_out : K = 900 = 225 chunks of 4 (exact) -----
    v8f acc2 = {};
    for (int ch = 0; ch < 225; ++ch) {
        int j0 = ch * 4 + kbase;
        int jpf = j0 + 32;                               // prefetch ahead
        if (jpf < NINF) __builtin_prefetch(&u_inf[jpf * NC + myrow], 0, 3);
        v2f a;
        a[0] = integ_term(u_inf[(j0    ) * NC + myrow], j0,     896);
        a[1] = integ_term(u_inf[(j0 + 1) * NC + myrow], j0 + 1, 896);
        acc2 = __builtin_amdgcn_wmma_f32_16x16x4_f32(
                   false, a, false, ones2, (short)0, acc2, false, false);
    }
    if (lane == 0) {
        #pragma unroll
        for (int p = 0; p < 8; ++p) ws[5 * NC + rowbase + p] = acc2[p];
    } else if (lane == 16) {
        #pragma unroll
        for (int p = 0; p < 8; ++p) ws[5 * NC + rowbase + 8 + p] = acc2[p];
    }
#else
    // Fallback: codegen-confirmed f16 WMMA with pre-scaling.
    int khalf = (lane >> 4) * 16;
    v16h ones;
    #pragma unroll
    for (int i = 0; i < 16; ++i) ones[i] = (_Float16)1.0f;
    const float S    = 1048576.0f;
    const float Sinv = 1.0f / S;

    v8f acc = {};
    #pragma unroll
    for (int ch = 0; ch < 4; ++ch) {
        v16h a;
        #pragma unroll
        for (int t = 0; t < 16; ++t) {
            int   j = ch * 32 + khalf + t;
            float u = (j < MATCH) ? u_zero[j * NC + myrow] : 0.0f;
            a[t] = (_Float16)(integ_term(u, j, 96) * S);
        }
        acc = __builtin_amdgcn_wmma_f32_16x16x32_f16(
                  false, a, false, ones, (short)0, acc, false, false);
    }
    if (lane == 0) {
        #pragma unroll
        for (int p = 0; p < 8; ++p) ws[4 * NC + rowbase + p] = acc[p] * Sinv;
    } else if (lane == 16) {
        #pragma unroll
        for (int p = 0; p < 8; ++p) ws[4 * NC + rowbase + 8 + p] = acc[p] * Sinv;
    }

    v8f acc2 = {};
    for (int ch = 0; ch < 29; ++ch) {
        v16h a;
        #pragma unroll
        for (int t = 0; t < 16; ++t) {
            int   j = ch * 32 + khalf + t;
            float u = (j < NINF) ? u_inf[j * NC + myrow] : 0.0f;
            a[t] = (_Float16)(integ_term(u, j, 896) * S);
        }
        acc2 = __builtin_amdgcn_wmma_f32_16x16x32_f16(
                   false, a, false, ones, (short)0, acc2, false, false);
    }
    if (lane == 0) {
        #pragma unroll
        for (int p = 0; p < 8; ++p) ws[5 * NC + rowbase + p] = acc2[p] * Sinv;
    } else if (lane == 16) {
        #pragma unroll
        for (int p = 0; p < 8; ++p) ws[5 * NC + rowbase + 8 + p] = acc2[p] * Sinv;
    }
#endif
}

// ---------------------------------------------------------------------------
// K3: eigenvalue update. Replicates the reference's [::-1] flip of lfunc_out
// along the energy axis via the cross-column workspace.
// ---------------------------------------------------------------------------
__global__ void __launch_bounds__(256)
delta_kernel(const float* __restrict__ init_energy,
             const float* __restrict__ ws, float* __restrict__ out) {
    int c = blockIdx.x * blockDim.x + threadIdx.x;
    if (c >= NC) return;
    int ie   = c / LDIM;
    int l    = c - ie * LDIM;
    int crev = (NE - 1 - ie) * LDIM + l;

    float lfi = ws[0 * NC + c];
    float ui0 = ws[3 * NC + c];
    float lfo = ws[1 * NC + crev] / ui0;
    float uzl = ws[2 * NC + c];
    float den = ws[4 * NC + c] / (uzl * uzl) + ws[5 * NC + c] / (ui0 * ui0);
    float delta = -(lfo - lfi) / den;
    out[c] = init_energy[ie] + delta;
}

// ---------------------------------------------------------------------------
extern "C" void kernel_launch(void* const* d_in, const int* in_sizes, int n_in,
                              void* d_out, int out_size, void* d_ws, size_t ws_size,
                              hipStream_t stream) {
    const float* init_energy = (const float*)d_in[0];
    float* out = (float*)d_out;
    float* ws  = (float*)d_ws;   // needs 6*NC floats = 288 KiB

    solver_kernel<<<NC / 128, 128, 0, stream>>>(init_energy, out, ws);
    integrate_kernel<<<(NC / 16) / 8, 256, 0, stream>>>(out, ws);   // 768 waves
    delta_kernel<<<NC / 256, 256, 0, stream>>>(init_energy, ws, out);
}
